// FullAttention_54812372632290
// MI455X (gfx1250) — matrix operand order repaired and checked by reading, not compiled, not run on hardware
//
#include <hip/hip_runtime.h>

#define BB 4
#define TT 2048
#define DIMM 1024
#define HH 16
#define DD 64
#define HDD (HH*DD)
#define BT (BB*TT)
#define SCALE 0.125f

typedef __bf16 bf16;
typedef __attribute__((ext_vector_type(16))) __bf16 v16bf;
typedef __attribute__((ext_vector_type(8)))  __bf16 v8bf;
typedef __attribute__((ext_vector_type(4)))  __bf16 v4bf;
typedef __attribute__((ext_vector_type(4)))  float  v4f;
typedef __attribute__((ext_vector_type(8)))  float  v8f;
typedef __attribute__((ext_vector_type(4)))  unsigned int u32x4;
typedef __attribute__((ext_vector_type(8)))  int    i32x8;
typedef __attribute__((ext_vector_type(4)))  int    i32x4;

__device__ __forceinline__ v16bf cat16(v8bf lo, v8bf hi) {
  v16bf r;
#pragma unroll
  for (int i = 0; i < 8; ++i) { r[i] = lo[i]; r[i + 8] = hi[i]; }
  return r;
}

// ---------------- TDM: 2D tile load (32 x 128 rows of bf16) into LDS ----------------
// Loads tile_dim0=32 elems (64B) per row, 128 rows, row stride = row_stride_elems,
// with LDS padding of 16B after every 64B row => LDS row stride of 40 bf16.
__device__ __forceinline__ void tdm_tile_2d(unsigned lds_byte_addr,
                                            const bf16* gsrc,
                                            int row_stride_elems) {
  unsigned long long ga = (unsigned long long)(size_t)gsrc;
  u32x4 g0;
  g0[0] = 1u;                                              // count=1, flags=0
  g0[1] = lds_byte_addr;                                   // lds_addr
  g0[2] = (unsigned)ga;                                    // global_addr[31:0]
  g0[3] = (unsigned)((ga >> 32) & 0x01ffffffull) | (2u << 30); // addr[56:32], type=2
  i32x8 g1;
  g1[0] = (int)((1u << 16) | (1u << 20) | (3u << 22) | (3u << 25));
  // data_size=2B | pad_enable | pad_interval=3 (16 dwords) | pad_amount=3 (4 dwords)
  g1[1] = (int)(32u << 16);      // atomic_barrier=0 ; tensor_dim0[15:0]=32
  g1[2] = (int)(128u << 16);     // tensor_dim0[31:16]=0 ; tensor_dim1[15:0]=128
  g1[3] = (int)(32u << 16);      // tensor_dim1[31:16]=0 ; tile_dim0=32
  g1[4] = (int)(128u);           // tile_dim1=128 ; tile_dim2=0
  g1[5] = row_stride_elems;      // tensor_dim0_stride[31:0]
  g1[6] = 0;                     // stride hi, dim1_stride lo
  g1[7] = 0;
  i32x4 z4 = {0, 0, 0, 0};
  i32x8 z8 = {0, 0, 0, 0, 0, 0, 0, 0};
  __builtin_amdgcn_tensor_load_to_lds(g0, g1, z4, z4, z8, 0);
}

// ---------------- fp32 -> bf16 conversion ----------------
__global__ void cvt_bf16(const float* __restrict__ src, bf16* __restrict__ dst, int n) {
  int i = (blockIdx.x * 256 + threadIdx.x) * 4;
  if (i < n) {
    v4f f = *(const v4f*)(src + i);
    v4bf o;
#pragma unroll
    for (int j = 0; j < 4; ++j) o[j] = (bf16)f[j];
    *(v4bf*)(dst + i) = o;
  }
}

// transpose convert: dst[c*R + r] = src[r*C + c]
__global__ void cvt_bf16_T(const float* __restrict__ src, bf16* __restrict__ dst, int R, int C) {
  int idx = blockIdx.x * 256 + threadIdx.x;
  if (idx < R * C) {
    int r = idx / C, c = idx % C;
    dst[(size_t)c * R + r] = (bf16)src[idx];
  }
}

// ---------------- tiled bf16 WMMA GEMM with TDM double-buffered staging ----------------
// C[M,N] = A[M,K] @ B[K,N], with B supplied transposed: Bt[N,K].
// mode 0: out = float row-major [M,N]
// mode 1: out = bf16 scattered to [B,H,T,D]  (row = b*T+t, col = h*D+d)
__global__ __launch_bounds__(256)
void gemm_bf16(const bf16* __restrict__ A, const bf16* __restrict__ Bt,
               void* __restrict__ out, int M, int N, int K, int mode) {
  __shared__ bf16 lA[2][128 * 40];
  __shared__ bf16 lB[2][128 * 40];
  const int tid  = threadIdx.x;
  const int lane = tid & 31;
  const int wave = tid >> 5;
  const int l16  = lane & 15;
  const int half = lane >> 4;
  const int m0 = blockIdx.y * 128;
  const int n0 = blockIdx.x * 128;
  const int wr = wave & 3;   // 4 waves along M -> 32 rows each
  const int wc = wave >> 2;  // 2 waves along N -> 64 cols each

  v8f acc[2][4];
#pragma unroll
  for (int mi = 0; mi < 2; ++mi)
#pragma unroll
    for (int ni = 0; ni < 4; ++ni) acc[mi][ni] = {};

  // prologue: wave0 issues TDM loads for the first k-tile into buffer 0
  if (wave == 0) {
    tdm_tile_2d((unsigned)(size_t)&lA[0][0], A + (size_t)m0 * K, K);
    tdm_tile_2d((unsigned)(size_t)&lB[0][0], Bt + (size_t)n0 * K, K);
  }

  const int nk = K >> 5;
  for (int it = 0; it < nk; ++it) {
    const int cur = it & 1;
    if (wave == 0) {
      if (it + 1 < nk) {
        // prefetch next tile into the other buffer (safe: previous iter's end
        // barrier guarantees everyone is done reading it), then wait until only
        // the 2 freshly issued TDM ops remain outstanding (in-order retirement
        // => current buffer's pair has landed in LDS).
        const int kn = (it + 1) << 5;
        tdm_tile_2d((unsigned)(size_t)&lA[cur ^ 1][0], A + (size_t)m0 * K + kn, K);
        tdm_tile_2d((unsigned)(size_t)&lB[cur ^ 1][0], Bt + (size_t)n0 * K + kn, K);
        __builtin_amdgcn_s_wait_tensorcnt(2);
      } else {
        __builtin_amdgcn_s_wait_tensorcnt(0);
      }
    }
    __syncthreads();

    v16bf af[2], bfrag[4];
#pragma unroll
    for (int mi = 0; mi < 2; ++mi) {
      // A 16x32 bf16 layout: lane row = l16, K chunks at (half?8:0) and +16
      const bf16* p = &lA[cur][0] + (wr * 32 + mi * 16 + l16) * 40 + half * 8;
      af[mi] = cat16(*(const v8bf*)p, *(const v8bf*)(p + 16));
    }
#pragma unroll
    for (int ni = 0; ni < 4; ++ni) {
      // B 32x16: lane col = l16, K chunks at (half?8:0) and +16
      const bf16* p = &lB[cur][0] + (wc * 64 + ni * 16 + l16) * 40 + half * 8;
      bfrag[ni] = cat16(*(const v8bf*)p, *(const v8bf*)(p + 16));
    }
#pragma unroll
    for (int mi = 0; mi < 2; ++mi)
#pragma unroll
      for (int ni = 0; ni < 4; ++ni)
        acc[mi][ni] = __builtin_amdgcn_wmma_f32_16x16x32_bf16(
            false, af[mi], false, bfrag[ni], (short)0, acc[mi][ni], false, false);
    __syncthreads();
  }

#pragma unroll
  for (int mi = 0; mi < 2; ++mi)
#pragma unroll
    for (int ni = 0; ni < 4; ++ni)
#pragma unroll
      for (int j = 0; j < 8; ++j) {
        int gr = m0 + wr * 32 + mi * 16 + j + half * 8;
        int gc = n0 + wc * 64 + ni * 16 + l16;
        float v = acc[mi][ni][j];
        if (mode == 0) {
          ((float*)out)[(size_t)gr * N + gc] = v;
        } else {
          int b = gr / TT, t = gr % TT;
          int h = gc / DD, d = gc % DD;
          ((bf16*)out)[(((size_t)(b * HH + h)) * TT + t) * DD + d] = (bf16)v;
        }
      }
}

// ---------------- RoPE on Q and K, in place, [B,H,T,D] bf16 ----------------
__global__ void rope_kernel(bf16* __restrict__ Q, bf16* __restrict__ Kk,
                            const float* __restrict__ cosp, const float* __restrict__ sinp) {
  int idx = blockIdx.x * 256 + threadIdx.x;   // over B*H*T*(D/2)
  int i  = idx & 31;                          // D/2 = 32
  int t  = (idx >> 5) & (TT - 1);
  int bh = idx >> 16;
  size_t base = ((size_t)bh * TT + t) * DD;
  float c = cosp[t * 32 + i], s = sinp[t * 32 + i];
  float q1 = (float)Q[base + i], q2 = (float)Q[base + i + 32];
  Q[base + i]      = (bf16)(q1 * c - q2 * s);
  Q[base + i + 32] = (bf16)(q2 * c + q1 * s);
  float k1 = (float)Kk[base + i], k2 = (float)Kk[base + i + 32];
  Kk[base + i]      = (bf16)(k1 * c - k2 * s);
  Kk[base + i + 32] = (bf16)(k2 * c + k1 * s);
}

// ---------------- flash attention (causal) ----------------
// grid: (T/128, B*H), 256 threads (8 waves, 16 q-rows per wave)
// O written as [B,T,H,D] bf16 (row-major [B*T, H*D] for the final GEMM)
__global__ __launch_bounds__(256)
void flash_attn(const bf16* __restrict__ Q, const bf16* __restrict__ Km,
                const bf16* __restrict__ V, bf16* __restrict__ O) {
  __shared__ bf16 lVt[64 * 40];        // V chunk transposed: Vt[d][kk]
  __shared__ bf16 lP[8 * 16 * 40];     // per-wave P scratch (16 x 32)
  const int tid  = threadIdx.x;
  const int lane = tid & 31;
  const int wave = tid >> 5;
  const int l16  = lane & 15;
  const int half = lane >> 4;
  const int bh = blockIdx.y;
  const int q0 = blockIdx.x * 128;
  const size_t base = (size_t)bh * TT * DD;

  // resident Q A-fragments (16 rows x 64, two K=32 chunks)
  v16bf aq[2];
  {
    const bf16* qp = Q + base + (size_t)(q0 + wave * 16 + l16) * DD;
#pragma unroll
    for (int dc = 0; dc < 2; ++dc) {
      int k0 = dc * 32 + half * 8;
      aq[dc] = cat16(*(const v8bf*)(qp + k0), *(const v8bf*)(qp + k0 + 16));
    }
  }

  v8f o[4];
#pragma unroll
  for (int nt = 0; nt < 4; ++nt) o[nt] = {};
  float mrow[8], lrow[8];
#pragma unroll
  for (int j = 0; j < 8; ++j) { mrow[j] = -__builtin_inff(); lrow[j] = 0.0f; }

  bf16* myP = lP + wave * 16 * 40;
  const int qrow0 = q0 + wave * 16 + half * 8;
  const int kend = q0 + 128;

  for (int kt = 0; kt < kend; kt += 32) {
    // cooperative stage of V chunk [32 x 64] -> transposed LDS Vt[64][32]
    {
      int i0 = tid * 8;
      int kk = i0 >> 6, d0 = i0 & 63;
      v8bf vv = *(const v8bf*)(V + base + (size_t)(kt + kk) * DD + d0);
#pragma unroll
      for (int e = 0; e < 8; ++e) lVt[(d0 + e) * 40 + kk] = vv[e];
    }
    __syncthreads();

    // S = Q K^T for 2 column tiles of 16 keys
    v8f s[2];
#pragma unroll
    for (int nt = 0; nt < 2; ++nt) {
      v8f acc = {};
#pragma unroll
      for (int dc = 0; dc < 2; ++dc) {
        const bf16* kp = Km + base + (size_t)(kt + nt * 16 + l16) * DD + dc * 32 + half * 8;
        v16bf bk = cat16(*(const v8bf*)kp, *(const v8bf*)(kp + 16));
        acc = __builtin_amdgcn_wmma_f32_16x16x32_bf16(
            false, aq[dc], false, bk, (short)0, acc, false, false);
      }
      s[nt] = acc;
    }

    // scale + causal mask
#pragma unroll
    for (int nt = 0; nt < 2; ++nt) {
      int col = kt + nt * 16 + l16;
#pragma unroll
      for (int j = 0; j < 8; ++j) {
        float v = s[nt][j] * SCALE;
        s[nt][j] = (col <= qrow0 + j) ? v : -__builtin_inff();
      }
    }

    // online softmax
    float alpha[8];
#pragma unroll
    for (int j = 0; j < 8; ++j) {
      float v = fmaxf(s[0][j], s[1][j]);
      v = fmaxf(v, __shfl_xor(v, 1));
      v = fmaxf(v, __shfl_xor(v, 2));
      v = fmaxf(v, __shfl_xor(v, 4));
      v = fmaxf(v, __shfl_xor(v, 8));
      float mn = fmaxf(mrow[j], v);
      alpha[j] = __expf(mrow[j] - mn);
      float p0 = __expf(s[0][j] - mn);
      float p1 = __expf(s[1][j] - mn);
      s[0][j] = p0; s[1][j] = p1;
      float rs = p0 + p1;
      rs += __shfl_xor(rs, 1);
      rs += __shfl_xor(rs, 2);
      rs += __shfl_xor(rs, 4);
      rs += __shfl_xor(rs, 8);
      lrow[j] = lrow[j] * alpha[j] + rs;
      mrow[j] = mn;
    }
#pragma unroll
    for (int nt = 0; nt < 4; ++nt)
#pragma unroll
      for (int j = 0; j < 8; ++j) o[nt][j] *= alpha[j];

    // transpose P (C-layout -> A-layout) through per-wave LDS scratch
#pragma unroll
    for (int nt = 0; nt < 2; ++nt)
#pragma unroll
      for (int j = 0; j < 8; ++j)
        myP[(j + half * 8) * 40 + nt * 16 + l16] = (bf16)s[nt][j];
    asm volatile("s_wait_dscnt 0x0" ::: "memory");
    v16bf pa;
    {
      const bf16* pp = myP + l16 * 40 + half * 8;
      pa = cat16(*(const v8bf*)pp, *(const v8bf*)(pp + 16));
    }

    // O += P @ V  (4 output d-tiles)
#pragma unroll
    for (int nt = 0; nt < 4; ++nt) {
      const bf16* vp = lVt + (nt * 16 + l16) * 40 + half * 8;
      v16bf bv = cat16(*(const v8bf*)vp, *(const v8bf*)(vp + 16));
      o[nt] = __builtin_amdgcn_wmma_f32_16x16x32_bf16(
          false, pa, false, bv, (short)0, o[nt], false, false);
    }
    __syncthreads();
  }

  // normalize and write [B,T,H,D]
  int b = bh / HH, h = bh % HH;
#pragma unroll
  for (int nt = 0; nt < 4; ++nt)
#pragma unroll
    for (int j = 0; j < 8; ++j) {
      int t = q0 + wave * 16 + j + half * 8;
      int d = nt * 16 + l16;
      float v = o[nt][j] / lrow[j];
      O[((size_t)(b * TT + t)) * HDD + h * DD + d] = (bf16)v;
    }
}

extern "C" void kernel_launch(void* const* d_in, const int* in_sizes, int n_in,
                              void* d_out, int out_size, void* d_ws, size_t ws_size,
                              hipStream_t stream) {
  const float* x    = (const float*)d_in[0];
  const float* cosp = (const float*)d_in[1];
  const float* sinp = (const float*)d_in[2];
  const float* Wq   = (const float*)d_in[3];
  const float* Wk   = (const float*)d_in[4];
  const float* Wv   = (const float*)d_in[5];
  const float* Wo   = (const float*)d_in[6];
  float* out = (float*)d_out;

  char* ws = (char*)d_ws;
  bf16* xb  = (bf16*)ws; ws += (size_t)BT * DIMM * 2;
  bf16* Wqt = (bf16*)ws; ws += (size_t)DIMM * HDD * 2;
  bf16* Wkt = (bf16*)ws; ws += (size_t)DIMM * HDD * 2;
  bf16* Wvt = (bf16*)ws; ws += (size_t)DIMM * HDD * 2;
  bf16* Wot = (bf16*)ws; ws += (size_t)HDD * DIMM * 2;
  bf16* Qb  = (bf16*)ws; ws += (size_t)BB * HH * TT * DD * 2;
  bf16* Kb  = (bf16*)ws; ws += (size_t)BB * HH * TT * DD * 2;
  bf16* Vb  = (bf16*)ws; ws += (size_t)BB * HH * TT * DD * 2;
  bf16* Ob  = (bf16*)ws; ws += (size_t)BT * HDD * 2;

  // 1) conversions
  {
    int n = BT * DIMM;
    cvt_bf16<<<n / (256 * 4), 256, 0, stream>>>(x, xb, n);
    int nw = DIMM * HDD;
    int g = (nw + 255) / 256;
    cvt_bf16_T<<<g, 256, 0, stream>>>(Wq, Wqt, DIMM, HDD);
    cvt_bf16_T<<<g, 256, 0, stream>>>(Wk, Wkt, DIMM, HDD);
    cvt_bf16_T<<<g, 256, 0, stream>>>(Wv, Wvt, DIMM, HDD);
    cvt_bf16_T<<<g, 256, 0, stream>>>(Wo, Wot, HDD, DIMM);
  }

  // 2) QKV projections (scatter to [B,H,T,D])
  {
    dim3 grid(HDD / 128, BT / 128);
    gemm_bf16<<<grid, 256, 0, stream>>>(xb, Wqt, (void*)Qb, BT, HDD, DIMM, 1);
    gemm_bf16<<<grid, 256, 0, stream>>>(xb, Wkt, (void*)Kb, BT, HDD, DIMM, 1);
    gemm_bf16<<<grid, 256, 0, stream>>>(xb, Wvt, (void*)Vb, BT, HDD, DIMM, 1);
  }

  // 3) RoPE on Q, K
  {
    int n = BB * HH * TT * (DD / 2);
    rope_kernel<<<n / 256, 256, 0, stream>>>(Qb, Kb, cosp, sinp);
  }

  // 4) causal flash attention
  {
    dim3 grid(TT / 128, BB * HH);
    flash_attn<<<grid, 256, 0, stream>>>(Qb, Kb, Vb, Ob);
  }

  // 5) output projection -> fp32 d_out
  {
    dim3 grid(DIMM / 128, BT / 128);
    gemm_bf16<<<grid, 256, 0, stream>>>(Ob, Wot, (void*)out, BT, DIMM, HDD, 0);
  }
}
